// WeatherPredictor_77283641524745
// MI455X (gfx1250) — matrix operand order repaired and checked by reading, not compiled, not run on hardware
//
#include <hip/hip_runtime.h>
#include <hip/hip_bf16.h>

// ============================================================================
// Spherical ViT forward for MI455X (gfx1250, wave32, WMMA).
// All matmuls use v_wmma_f32_16x16x32_bf16 (bf16 in, fp32 accumulate).
// Attention is a "transposed" flash attention: S^T = K @ Q^T, O^T = V^T @ P^T,
// which makes softmax row-stats per-lane (one shfl_xor(16) to merge halves)
// and converts P^T(D-frag) -> B-frag with register shuffles only (no LDS).
// GEMMs use 32x64 tiles per wave (8 WMMAs per k-step, ~21 FLOP/B from cache)
// plus global_prefetch hints for the next k-chunk.
// ============================================================================

typedef __bf16 bf16;
typedef __attribute__((ext_vector_type(16))) __bf16 v16bf;
typedef __attribute__((ext_vector_type(8)))  __bf16 v8bf;
typedef __attribute__((ext_vector_type(8)))  float  v8f;
typedef __attribute__((ext_vector_type(4)))  float  v4f;

#define D_MODEL 384
#define NHEAD   6
#define HDIM    64
#define NTOK    4096
#define BATCH   2
#define ROWS    (BATCH * NTOK)     // 8192
#define DHID    1536
#define QSCALE  0.125f             // 64^-0.5

static __device__ __forceinline__ v8f zero8() {
  v8f z = {0.f,0.f,0.f,0.f,0.f,0.f,0.f,0.f};
  return z;
}

static __device__ __forceinline__ v8f wmma_bf16(v16bf a, v16bf b, v8f c) {
  return __builtin_amdgcn_wmma_f32_16x16x32_bf16(false, a, false, b, (short)0, c, false, false);
}

// A-fragment 16x32 (rows row0..row0+15, contraction k..k+31) from row-major bf16
// lane = m + 16h : elems 0..15 = K[k+16h .. +15]
static __device__ __forceinline__ v16bf load_afrag(const bf16* base, int ld, int row0, int k, int lane) {
  int m = lane & 15, h = lane >> 4;
  const bf16* p = base + (size_t)(row0 + m) * ld + k + 16 * h;
  union { v16bf v; v8bf h2[2]; } u;
  u.h2[0] = *(const v8bf*)p;
  u.h2[1] = *(const v8bf*)(p + 8);
  return u.v;
}

// B-fragment 32x16 (cols col0..col0+15, contraction k..k+31) from row-major
// "transposed weight" bf16 [Ncols][K]: lane = n + 16h holds K[k+16h .. +15].
static __device__ __forceinline__ v16bf load_bfrag(const bf16* base, int ld, int col0, int k, int lane) {
  int n = lane & 15, h = lane >> 4;
  const bf16* p = base + (size_t)(col0 + n) * ld + k + 16 * h;
  union { v16bf v; v8bf h2[2]; } u;
  u.h2[0] = *(const v8bf*)p;
  u.h2[1] = *(const v8bf*)(p + 8);
  return u.v;
}

// ---------------------------------------------------------------------------
// Small setup kernels
// ---------------------------------------------------------------------------
__global__ void transw_kernel(const float* __restrict__ w, bf16* __restrict__ wt, int K, int N) {
  int i = blockIdx.x * 256 + threadIdx.x;          // i = n*K + k
  if (i >= K * N) return;
  int n = i / K, k = i - n * K;
  wt[i] = (bf16)w[(size_t)k * N + n];
}

__global__ void logw_kernel(const float* __restrict__ gw, float* __restrict__ lw) {
  int i = blockIdx.x * 256 + threadIdx.x;
  if (i < NTOK) lw[i] = logf(gw[i]);
}

__global__ void embed_kernel(const float* __restrict__ x, const float* __restrict__ pe_w,
                             const float* __restrict__ pe_b, const float* __restrict__ pos,
                             float* __restrict__ t) {
  int i = blockIdx.x * 256 + threadIdx.x;
  if (i >= ROWS * D_MODEL) return;
  int d = i % D_MODEL;
  int bn = i / D_MODEL;
  int n = bn % NTOK;
  t[i] = x[bn] * pe_w[d] + pe_b[d] + pos[(size_t)n * D_MODEL + d];
}

// One wave per row: LayerNorm fp32 -> bf16
__global__ __launch_bounds__(256) void ln_kernel(const float* __restrict__ t,
                                                 const float* __restrict__ s,
                                                 const float* __restrict__ b,
                                                 bf16* __restrict__ h) {
  int lane = threadIdx.x & 31, w = threadIdx.x >> 5;
  int row = blockIdx.x * 8 + w;
  const float* x = t + (size_t)row * D_MODEL;
  float vals[12], sum = 0.f, sq = 0.f;
#pragma unroll
  for (int i = 0; i < 12; i++) { float v = x[lane + i * 32]; vals[i] = v; sum += v; sq += v * v; }
#pragma unroll
  for (int m = 16; m >= 1; m >>= 1) { sum += __shfl_xor(sum, m, 32); sq += __shfl_xor(sq, m, 32); }
  float mean = sum * (1.f / D_MODEL);
  float var  = sq * (1.f / D_MODEL) - mean * mean;
  float rstd = rsqrtf(var + 1e-5f);
  bf16* o = h + (size_t)row * D_MODEL;
#pragma unroll
  for (int i = 0; i < 12; i++) {
    int d = lane + i * 32;
    o[d] = (bf16)((vals[i] - mean) * rstd * s[d] + b[d]);
  }
}

// Final LN + head projection (dot over D) -> out[b*4096+n]
__global__ __launch_bounds__(256) void final_head_kernel(const float* __restrict__ t,
                                                         const float* __restrict__ s,
                                                         const float* __restrict__ b,
                                                         const float* __restrict__ hw,
                                                         const float* __restrict__ hb,
                                                         float* __restrict__ out) {
  int lane = threadIdx.x & 31, w = threadIdx.x >> 5;
  int row = blockIdx.x * 8 + w;
  const float* x = t + (size_t)row * D_MODEL;
  float vals[12], sum = 0.f, sq = 0.f;
#pragma unroll
  for (int i = 0; i < 12; i++) { float v = x[lane + i * 32]; vals[i] = v; sum += v; sq += v * v; }
#pragma unroll
  for (int m = 16; m >= 1; m >>= 1) { sum += __shfl_xor(sum, m, 32); sq += __shfl_xor(sq, m, 32); }
  float mean = sum * (1.f / D_MODEL);
  float rstd = rsqrtf(sq * (1.f / D_MODEL) - mean * mean + 1e-5f);
  float acc = 0.f;
#pragma unroll
  for (int i = 0; i < 12; i++) {
    int d = lane + i * 32;
    acc += ((vals[i] - mean) * rstd * s[d] + b[d]) * hw[d];
  }
#pragma unroll
  for (int m = 16; m >= 1; m >>= 1) acc += __shfl_xor(acc, m, 32);
  if (lane == 0) out[row] = acc + hb[0];
}

// ---------------------------------------------------------------------------
// WMMA GEMM: C[M,N] = A[M,K] @ Wt^T,  A bf16 row-major (ld=K), Wt bf16 [N][K].
// Block = 8 waves; wave computes a 32(M)x64(N) tile: 2 A-frags x 4 B-frags,
// 8 WMMAs per 32-wide k-step, with prefetch of the next k-chunk.
// MODE 0: scatter qkv -> Q (scaled), K, V^T   (N=1152)
// MODE 1: t[row][col] += acc + bias[col]      (N=384, fp32 residual update)
// MODE 2: u[row][col] = gelu(acc + bias[col]) (N=1536, bf16 out)
// ---------------------------------------------------------------------------
template <int MODE>
__global__ __launch_bounds__(256) void gemm_kernel(const bf16* __restrict__ A,
                                                   const bf16* __restrict__ Wt,
                                                   const float* __restrict__ bias,
                                                   float* __restrict__ tres,
                                                   bf16* __restrict__ outbf,
                                                   bf16* __restrict__ Qo,
                                                   bf16* __restrict__ Ko,
                                                   bf16* __restrict__ Vto,
                                                   int Kdim) {
  int lane = threadIdx.x & 31, w = threadIdx.x >> 5;
  int m0 = blockIdx.x * 256 + w * 32;
  int n0 = blockIdx.y * 64;
  v8f acc[2][4];
#pragma unroll
  for (int s = 0; s < 2; s++)
#pragma unroll
    for (int ns = 0; ns < 4; ns++) acc[s][ns] = zero8();

  int ml = lane & 15, hl = lane >> 4;
  for (int k = 0; k < Kdim; k += 32) {
    if (k + 32 < Kdim) {
      // prefetch next k-chunk: A rows of this wave tile + a spread of B panel cols
      __builtin_prefetch(A  + (size_t)(m0 + ml + 16 * hl) * Kdim + k + 32, 0, 3);
      __builtin_prefetch(Wt + (size_t)(n0 + ml * 4 + hl * 2) * Kdim + k + 32, 0, 3);
    }
    v16bf a0 = load_afrag(A, Kdim, m0,      k, lane);
    v16bf a1 = load_afrag(A, Kdim, m0 + 16, k, lane);
#pragma unroll
    for (int ns = 0; ns < 4; ns++) {
      v16bf bfr = load_bfrag(Wt, Kdim, n0 + ns * 16, k, lane);
      acc[0][ns] = wmma_bf16(a0, bfr, acc[0][ns]);
      acc[1][ns] = wmma_bf16(a1, bfr, acc[1][ns]);
    }
  }

#pragma unroll
  for (int s = 0; s < 2; s++) {
#pragma unroll
    for (int ns = 0; ns < 4; ns++) {
      int col = n0 + ns * 16 + ml;
#pragma unroll
      for (int r = 0; r < 8; r++) {
        int row = m0 + s * 16 + 8 * hl + r;
        float v = acc[s][ns][r];
        if (MODE == 0) {
          int part = col / D_MODEL;             // 0=q 1=k 2=v
          int cm   = col - part * D_MODEL;
          int head = cm >> 6;
          int hd   = cm & 63;
          int bb   = row >> 12;                 // / 4096
          int q    = row & 4095;
          size_t bh = (size_t)(bb * NHEAD + head);
          if (part == 0)      Qo [(bh * NTOK + q) * HDIM + hd] = (bf16)(v * QSCALE);
          else if (part == 1) Ko [(bh * NTOK + q) * HDIM + hd] = (bf16)v;
          else                Vto[(bh * HDIM + hd) * NTOK + q] = (bf16)v;
        } else if (MODE == 1) {
          tres[(size_t)row * D_MODEL + col] += v + bias[col];
        } else {
          float xg = v + bias[col];
          float g = 0.5f * xg * (1.0f + erff(xg * 0.70710678118f));
          outbf[(size_t)row * DHID + col] = (bf16)g;
        }
      }
    }
  }
}

// ---------------------------------------------------------------------------
// Transposed flash attention. Grid: (N/128, NHEAD, BATCH), 256 threads.
// Wave handles 16 queries; loops over all 4096 keys in steps of 32.
//   S^T(16k x 16q) = K_rows(A) @ Q_rows(B)   [2 hd-chunk WMMAs per key-group]
//   O^T(64hd x 16q) accumulated as Vt_rows(A) @ P^T(B), 4 WMMAs per step.
// ---------------------------------------------------------------------------
__global__ __launch_bounds__(256) void attn_kernel(const bf16* __restrict__ Q,
                                                   const bf16* __restrict__ Km,
                                                   const bf16* __restrict__ Vt,
                                                   const float* __restrict__ lw,
                                                   bf16* __restrict__ attn_out) {
  int lane = threadIdx.x & 31, w = threadIdx.x >> 5;
  int q0   = blockIdx.x * 128 + w * 16;
  int head = blockIdx.y;
  int bb   = blockIdx.z;
  size_t bh = (size_t)(bb * NHEAD + head);
  const bf16* Qh  = Q  + bh * NTOK * HDIM;
  const bf16* Kh  = Km + bh * NTOK * HDIM;
  const bf16* Vth = Vt + bh * HDIM * NTOK;

  int n = lane & 15, h = lane >> 4;

  // Q as B-fragments (held for the whole key loop): hd chunks [0,32) and [32,64)
  v16bf qb0 = load_bfrag(Qh, HDIM, q0, 0,  lane);
  v16bf qb1 = load_bfrag(Qh, HDIM, q0, 32, lane);

  v8f acc[4] = {zero8(), zero8(), zero8(), zero8()};  // O^T, hd groups of 16
  float m_run = -1e30f, l_run = 0.f;                  // stats for query q0+n

  for (int k0 = 0; k0 < NTOK; k0 += 32) {
    if (k0 + 32 < NTOK) {
      __builtin_prefetch(Kh  + (size_t)(k0 + 32 + lane) * HDIM, 0, 3);
      __builtin_prefetch(Vth + (size_t)(lane * 2) * NTOK + k0 + 32, 0, 3);
    }
    // ---- S^T fragments: rows = keys, cols = queries
    v8f s0 = zero8(), s1 = zero8();
    v16bf ka;
    ka = load_afrag(Kh, HDIM, k0,      0,  lane); s0 = wmma_bf16(ka, qb0, s0);
    ka = load_afrag(Kh, HDIM, k0,      32, lane); s0 = wmma_bf16(ka, qb1, s0);
    ka = load_afrag(Kh, HDIM, k0 + 16, 0,  lane); s1 = wmma_bf16(ka, qb0, s1);
    ka = load_afrag(Kh, HDIM, k0 + 16, 32, lane); s1 = wmma_bf16(ka, qb1, s1);

    // ---- per-key log-quadrature bias (key = row of S^T)
    {
      const float* p0 = lw + k0 + 8 * h;
      const float* p1 = lw + k0 + 16 + 8 * h;
      v4f a0 = *(const v4f*)p0, a1 = *(const v4f*)(p0 + 4);
      v4f b0 = *(const v4f*)p1, b1 = *(const v4f*)(p1 + 4);
#pragma unroll
      for (int r = 0; r < 4; r++) {
        s0[r] += a0[r]; s0[4 + r] += a1[r];
        s1[r] += b0[r]; s1[4 + r] += b1[r];
      }
    }

    // ---- online softmax stats for this query (column): in-lane + 1 shuffle
    float mx = -1e30f;
#pragma unroll
    for (int r = 0; r < 8; r++) mx = fmaxf(mx, fmaxf(s0[r], s1[r]));
    mx = fmaxf(mx, __shfl_xor(mx, 16, 32));
    float mnew = fmaxf(m_run, mx);
    float fac  = __expf(m_run - mnew);
    float rs   = 0.f;
#pragma unroll
    for (int r = 0; r < 8; r++) {
      float p0 = __expf(s0[r] - mnew); s0[r] = p0; rs += p0;
      float p1 = __expf(s1[r] - mnew); s1[r] = p1; rs += p1;
    }
    rs += __shfl_xor(rs, 16, 32);
    l_run = l_run * fac + rs;
    m_run = mnew;
#pragma unroll
    for (int g = 0; g < 4; g++) acc[g] *= fac;

    // ---- P^T D-frag -> B-frag (keys k0..k0+31 along K) via cross-half shuffles
    union { v16bf v; __bf16 e[16]; } pb;
#pragma unroll
    for (int r = 0; r < 8; r++) {
      float x0 = __shfl_xor(s0[r], 16, 32);   // partner's group-0 probs
      float x1 = __shfl_xor(s1[r], 16, 32);   // partner's group-1 probs
      float lo = (h == 0) ? s0[r] : x1;       // h=0: keys r    ; h=1: keys 16+r
      float hi = (h == 0) ? x0    : s1[r];    // h=0: keys 8+r  ; h=1: keys 24+r
      pb.e[r]     = (bf16)lo;
      pb.e[8 + r] = (bf16)hi;
    }

    // ---- O^T += V^T @ P^T  (4 hd groups of 16 rows)
#pragma unroll
    for (int g = 0; g < 4; g++) {
      v16bf va = load_afrag(Vth, NTOK, g * 16, k0, lane);
      acc[g] = wmma_bf16(va, pb.v, acc[g]);
    }
  }

  // ---- normalize and store O[q][hd] as bf16 rows of attn_out [ROWS][D]
  float inv = 1.0f / l_run;
  size_t row = (size_t)(bb * NTOK + q0 + n);
  bf16* o = attn_out + row * D_MODEL + head * HDIM;
#pragma unroll
  for (int g = 0; g < 4; g++)
#pragma unroll
    for (int r = 0; r < 8; r++)
      o[g * 16 + 8 * h + r] = (bf16)(acc[g][r] * inv);
}

// ---------------------------------------------------------------------------
// Host launcher
// ---------------------------------------------------------------------------
extern "C" void kernel_launch(void* const* d_in, const int* in_sizes, int n_in,
                              void* d_out, int out_size, void* d_ws, size_t ws_size,
                              hipStream_t stream) {
  const float* x      = (const float*)d_in[0];
  const float* pe_w   = (const float*)d_in[1];
  const float* pe_b   = (const float*)d_in[2];
  const float* pos    = (const float*)d_in[3];
  const float* ln1_s  = (const float*)d_in[4];
  const float* ln1_b  = (const float*)d_in[5];
  const float* qkv_w  = (const float*)d_in[6];
  const float* proj_w = (const float*)d_in[7];
  const float* proj_b = (const float*)d_in[8];
  const float* ln2_s  = (const float*)d_in[9];
  const float* ln2_b  = (const float*)d_in[10];
  const float* mlp_w1 = (const float*)d_in[11];
  const float* mlp_b1 = (const float*)d_in[12];
  const float* mlp_w2 = (const float*)d_in[13];
  const float* mlp_b2 = (const float*)d_in[14];
  const float* lnf_s  = (const float*)d_in[15];
  const float* lnf_b  = (const float*)d_in[16];
  const float* head_w = (const float*)d_in[17];
  const float* head_b = (const float*)d_in[18];
  const float* gw     = (const float*)d_in[19];

  // ---- workspace carve (256B aligned)
  char* p = (char*)d_ws;
  auto alloc = [&](size_t bytes) -> void* {
    void* r = (void*)p;
    p += (bytes + 255) & ~(size_t)255;
    return r;
  };
  float* t     = (float*)alloc((size_t)ROWS * D_MODEL * 4);
  bf16*  hbuf  = (bf16*) alloc((size_t)ROWS * D_MODEL * 2);
  bf16*  Qb    = (bf16*) alloc((size_t)BATCH * NHEAD * NTOK * HDIM * 2);
  bf16*  Kb    = (bf16*) alloc((size_t)BATCH * NHEAD * NTOK * HDIM * 2);
  bf16*  Vtb   = (bf16*) alloc((size_t)BATCH * NHEAD * HDIM * NTOK * 2);
  bf16*  attn  = (bf16*) alloc((size_t)ROWS * D_MODEL * 2);
  bf16*  u     = (bf16*) alloc((size_t)ROWS * DHID * 2);
  bf16*  qkvwt = (bf16*) alloc((size_t)4 * 3 * D_MODEL * D_MODEL * 2);
  bf16*  projwt= (bf16*) alloc((size_t)4 * D_MODEL * D_MODEL * 2);
  bf16*  m1wt  = (bf16*) alloc((size_t)4 * DHID * D_MODEL * 2);
  bf16*  m2wt  = (bf16*) alloc((size_t)4 * D_MODEL * DHID * 2);
  float* lw    = (float*)alloc((size_t)NTOK * 4);

  // ---- one-time per call: weight transposes (fp32 -> bf16 [N][K]) + log weights
  for (int i = 0; i < 4; i++) {
    transw_kernel<<<(D_MODEL * 3 * D_MODEL + 255) / 256, 256, 0, stream>>>(
        qkv_w + (size_t)i * D_MODEL * 3 * D_MODEL, qkvwt + (size_t)i * 3 * D_MODEL * D_MODEL,
        D_MODEL, 3 * D_MODEL);
    transw_kernel<<<(D_MODEL * D_MODEL + 255) / 256, 256, 0, stream>>>(
        proj_w + (size_t)i * D_MODEL * D_MODEL, projwt + (size_t)i * D_MODEL * D_MODEL,
        D_MODEL, D_MODEL);
    transw_kernel<<<(D_MODEL * DHID + 255) / 256, 256, 0, stream>>>(
        mlp_w1 + (size_t)i * D_MODEL * DHID, m1wt + (size_t)i * DHID * D_MODEL,
        D_MODEL, DHID);
    transw_kernel<<<(DHID * D_MODEL + 255) / 256, 256, 0, stream>>>(
        mlp_w2 + (size_t)i * DHID * D_MODEL, m2wt + (size_t)i * D_MODEL * DHID,
        DHID, D_MODEL);
  }
  logw_kernel<<<(NTOK + 255) / 256, 256, 0, stream>>>(gw, lw);

  // ---- patch embed
  embed_kernel<<<(ROWS * D_MODEL + 255) / 256, 256, 0, stream>>>(x, pe_w, pe_b, pos, t);

  // ---- transformer layers
  for (int i = 0; i < 4; i++) {
    ln_kernel<<<ROWS / 8, 256, 0, stream>>>(t, ln1_s + i * D_MODEL, ln1_b + i * D_MODEL, hbuf);
    gemm_kernel<0><<<dim3(ROWS / 256, (3 * D_MODEL) / 64), 256, 0, stream>>>(
        hbuf, qkvwt + (size_t)i * 3 * D_MODEL * D_MODEL, nullptr, nullptr, nullptr,
        Qb, Kb, Vtb, D_MODEL);
    attn_kernel<<<dim3(NTOK / 128, NHEAD, BATCH), 256, 0, stream>>>(Qb, Kb, Vtb, lw, attn);
    gemm_kernel<1><<<dim3(ROWS / 256, D_MODEL / 64), 256, 0, stream>>>(
        attn, projwt + (size_t)i * D_MODEL * D_MODEL, proj_b + i * D_MODEL, t, nullptr,
        nullptr, nullptr, nullptr, D_MODEL);
    ln_kernel<<<ROWS / 8, 256, 0, stream>>>(t, ln2_s + i * D_MODEL, ln2_b + i * D_MODEL, hbuf);
    gemm_kernel<2><<<dim3(ROWS / 256, DHID / 64), 256, 0, stream>>>(
        hbuf, m1wt + (size_t)i * DHID * D_MODEL, mlp_b1 + i * DHID, nullptr, u,
        nullptr, nullptr, nullptr, D_MODEL);
    gemm_kernel<1><<<dim3(ROWS / 256, D_MODEL / 64), 256, 0, stream>>>(
        u, m2wt + (size_t)i * D_MODEL * DHID, mlp_b2 + i * D_MODEL, t, nullptr,
        nullptr, nullptr, nullptr, DHID);
  }

  // ---- final LN + head
  final_head_kernel<<<ROWS / 8, 256, 0, stream>>>(t, lnf_s, lnf_b, head_w, head_b, (float*)d_out);
}
